// PointerProbGenerator_13065290514978
// MI455X (gfx1250) — matrix operand order repaired and checked by reading, not compiled, not run on hardware
//
#include <hip/hip_runtime.h>
#include <hip/hip_bf16.h>
#include <math.h>

// ---------------------------------------------------------------------------
// PointerProbGenerator for MI455X (gfx1250, wave32, WMMA bf16 16x16x32)
// q = query@Wq + bq ; k = key@Wk + bk ; S = q·kᵀ ; softmax(mask(10·tanh(S)/√d))
// B=16, Lq=512, Lk=4096, D=128
// ---------------------------------------------------------------------------

typedef __attribute__((ext_vector_type(16))) __bf16 v16bf;
typedef __attribute__((ext_vector_type(8)))  float  v8f;
typedef __attribute__((ext_vector_type(4)))  float  v4f;

#define NB   16
#define LQ   512
#define LK   4096
#define DM   128
#define QROWS (NB * LQ)     // 8192
#define KROWS (NB * LK)     // 65536
#define LKS  4100           // LDS row stride (floats): 4100 % 64 = 4 -> bank skew

// Native transcendental tanh on gfx1250 (1 TRANS op, co-executes with VALU)
#if __has_builtin(__builtin_amdgcn_tanhf)
  #define FAST_TANH(x) __builtin_amdgcn_tanhf(x)
#elif __has_builtin(__builtin_amdgcn_tanh_f32)
  #define FAST_TANH(x) __builtin_amdgcn_tanh_f32(x)
#else
  #define FAST_TANH(x) tanhf(x)
#endif

// A operand (16x32 bf16 chunk): lane = lh*16 + ln, row M = ln.
// elems 0-7  = row[k0 + lh*16 .. +8]
// elems 8-15 = row[k0 + lh*16 + 8 .. +8]
__device__ __forceinline__ v16bf load_a_bf16(const __bf16* row, int k0, int lh) {
  union { v16bf v; uint4 q[2]; } r;
  r.q[0] = *(const uint4*)(row + k0 + lh * 16);
  r.q[1] = *(const uint4*)(row + k0 + lh * 16 + 8);
  return r.v;
}

// Same A layout but source is f32, converted with native v_cvt_pk_bf16_f32.
__device__ __forceinline__ v16bf load_a_f32(const float* row, int k0, int lh) {
  v16bf r;
  const float* p0 = row + k0 + lh * 16;
  const float* p1 = row + k0 + lh * 16 + 8;
#pragma unroll
  for (int i = 0; i < 8; ++i) {
    r[i]     = (__bf16)p0[i];
    r[8 + i] = (__bf16)p1[i];
  }
  return r;
}

// B operand (32x16 bf16 chunk): lane holds column n = ln, k = lh*16 + {0..15},
// i.e. 32 contiguous bytes of the (row-major) source row at k offset k0+lh*16.
__device__ __forceinline__ v16bf load_b_bf16(const __bf16* p) {
  union { v16bf v; uint4 q[2]; } r;
  const uint4* s = (const uint4*)p;
  r.q[0] = s[0];
  r.q[1] = s[1];
  return r.v;
}

// ---------------------------------------------------------------------------
// Kernel 0: transpose + convert Wq, Wk (f32 [d][e]) -> WT bf16 [e][d]
// ---------------------------------------------------------------------------
__global__ __launch_bounds__(256) void wtrans_kernel(
    const float* __restrict__ Wq, const float* __restrict__ Wk,
    __bf16* __restrict__ WqT, __bf16* __restrict__ WkT) {
  int j = blockIdx.x * 256 + threadIdx.x;      // 0 .. 32767
  int which = j >> 14;
  int idx = j & 16383;
  int e = idx >> 7, d = idx & 127;
  const float* W = which ? Wk : Wq;
  __bf16* WT = which ? WkT : WqT;
  WT[e * DM + d] = (__bf16)W[d * DM + e];
}

// ---------------------------------------------------------------------------
// Kernel 1: projections. Block = 4 waves; blocks [0,512) -> q, [512,4608) -> k.
// Each block produces a 16-row x 128-col bf16 tile of q_proj / k_proj.
// ---------------------------------------------------------------------------
__global__ __launch_bounds__(128) void proj_kernel(
    const float* __restrict__ query, const float* __restrict__ key,
    const __bf16* __restrict__ WqT, const __bf16* __restrict__ WkT,
    const float* __restrict__ bq, const float* __restrict__ bk,
    __bf16* __restrict__ qp, __bf16* __restrict__ kp) {
  int blk = blockIdx.x;
  const float* src; const __bf16* WT; const float* bias; __bf16* dst;
  long rowBase;
  if (blk < (QROWS / 16)) {
    src = query; WT = WqT; bias = bq; dst = qp; rowBase = (long)blk * 16;
  } else {
    src = key;   WT = WkT; bias = bk; dst = kp; rowBase = (long)(blk - QROWS / 16) * 16;
  }
  int lane = threadIdx.x & 31;
  int wave = threadIdx.x >> 5;
  int lh = lane >> 4;      // which half of the wave
  int ln = lane & 15;

  const float* arow = src + (rowBase + ln) * DM;
  v16bf a[4];
#pragma unroll
  for (int ks = 0; ks < 4; ++ks) a[ks] = load_a_f32(arow, ks * 32, lh);

#pragma unroll
  for (int t = 0; t < 2; ++t) {
    int n = (wave * 2 + t) * 16 + ln;          // output column (and WT row)
    const __bf16* brow = WT + n * DM;
    v8f c = {};
#pragma unroll
    for (int ks = 0; ks < 4; ++ks) {
      v16bf bm = load_b_bf16(brow + ks * 32 + lh * 16);
      c = __builtin_amdgcn_wmma_f32_16x16x32_bf16(false, a[ks], false, bm,
                                                  (short)0, c, false, false);
    }
    float bv = bias[n];
#pragma unroll
    for (int v = 0; v < 8; ++v) {
      int m = v + lh * 8;                      // C/D layout: lane half selects row block
      dst[(rowBase + m) * DM + n] = (__bf16)(c[v] + bv);
    }
  }
}

// ---------------------------------------------------------------------------
// Kernel 2: fused score + tanh/clip/mask + softmax.
// Block (8 waves) owns one (b, 16-row q tile) x all 4096 k columns.
// tanh bounds scores to ±0.884 -> exp() is safe without max subtraction.
// Mask reads and output writes are non-temporal (streamed once) so the
// 32x-reused k_proj tiles stay resident in the 192 MB L2.
// ---------------------------------------------------------------------------
__global__ __launch_bounds__(256) void score_softmax_kernel(
    const __bf16* __restrict__ qp, const __bf16* __restrict__ kp,
    const int* __restrict__ mask, float* __restrict__ out) {
  __shared__ float smem[16 * LKS];             // exp-scores, 262,400 B
  __shared__ float wsum[8][16];                // per-wave partial row sums

  int blk = blockIdx.x;                        // 0..511 = b*32 + qtile
  int b = blk >> 5;
  long rowBase = (long)blk * 16;               // flattened q row = b*512 + qtile*16
  int lane = threadIdx.x & 31;
  int wave = threadIdx.x >> 5;
  int lh = lane >> 4;
  int ln = lane & 15;

  // ---- phase 1: WMMA score tiles -> exp -> LDS, partial sums in registers
  const __bf16* arow = qp + (rowBase + ln) * DM;
  v16bf a[4];
#pragma unroll
  for (int ks = 0; ks < 4; ++ks) a[ks] = load_a_bf16(arow, ks * 32, lh);

  const long kbase = (long)b * LK;
  const float scale = 0.88388347648318440f;    // 10 / sqrt(128)
  float psum[8];
#pragma unroll
  for (int v = 0; v < 8; ++v) psum[v] = 0.0f;

  for (int t = 0; t < 32; ++t) {
    int n = wave * 512 + t * 16 + ln;          // local k column; also B-operand row
    const __bf16* brow = kp + (kbase + n) * DM;
    v8f c = {};
#pragma unroll
    for (int ks = 0; ks < 4; ++ks) {
      v16bf bm = load_b_bf16(brow + ks * 32 + lh * 16);
      c = __builtin_amdgcn_wmma_f32_16x16x32_bf16(false, a[ks], false, bm,
                                                  (short)0, c, false, false);
    }
    const int* mcol = mask + rowBase * LK + n;
#pragma unroll
    for (int v = 0; v < 8; ++v) {
      int m = v + lh * 8;
      float s = scale * FAST_TANH(c[v]);
      int mk = __builtin_nontemporal_load(mcol + (long)m * LK);
      float e = mk ? __expf(s) : 0.0f;         // masked -> exp(-1e9) == 0
      smem[m * LKS + n] = e;
      psum[v] += e;
    }
  }
  // reduce partial sums across the 16 lanes of each half (wave32 shuffles)
#pragma unroll
  for (int o = 8; o >= 1; o >>= 1) {
#pragma unroll
    for (int v = 0; v < 8; ++v) psum[v] += __shfl_xor(psum[v], o, 32);
  }
  if (ln == 0) {
#pragma unroll
    for (int v = 0; v < 8; ++v) wsum[wave][lh * 8 + v] = psum[v];
  }
  __syncthreads();

  // ---- phase 2: normalize + coalesced float4 non-temporal write-out
#pragma unroll
  for (int rr = 0; rr < 2; ++rr) {
    int m = wave * 2 + rr;
    float tot = 0.0f;
#pragma unroll
    for (int w = 0; w < 8; ++w) tot += wsum[w][m];
    float inv = 1.0f / tot;
    const v4f* row4 = (const v4f*)(smem + m * LKS);           // 16-B aligned
    v4f* orow4 = (v4f*)(out + (rowBase + m) * (long)LK);      // 16-B aligned
    for (int i = lane; i < LK / 4; i += 32) {
      v4f e4 = row4[i] * inv;
      __builtin_nontemporal_store(e4, orow4 + i);
    }
  }
}

// ---------------------------------------------------------------------------
extern "C" void kernel_launch(void* const* d_in, const int* in_sizes, int n_in,
                              void* d_out, int out_size, void* d_ws, size_t ws_size,
                              hipStream_t stream) {
  const float* query = (const float*)d_in[0];   // [16,512,128]
  const float* key   = (const float*)d_in[1];   // [16,4096,128]
  const int*   mask  = (const int*)d_in[2];     // [16,512,4096]
  const float* Wq    = (const float*)d_in[3];   // [128,128]
  const float* bq    = (const float*)d_in[4];   // [128]
  const float* Wk    = (const float*)d_in[5];   // [128,128]
  const float* bk    = (const float*)d_in[6];   // [128]
  float* out = (float*)d_out;                   // [16,512,4096]

  __bf16* qp  = (__bf16*)d_ws;                  // bf16 [8192][128]
  __bf16* kp  = qp + (size_t)QROWS * DM;        // bf16 [65536][128]
  __bf16* WqT = kp + (size_t)KROWS * DM;        // bf16 [128][128]
  __bf16* WkT = WqT + DM * DM;                  // bf16 [128][128]

  wtrans_kernel<<<128, 256, 0, stream>>>(Wq, Wk, WqT, WkT);
  proj_kernel<<<(QROWS / 16) + (KROWS / 16), 128, 0, stream>>>(
      query, key, WqT, WkT, bq, bk, qp, kp);
  score_softmax_kernel<<<NB * (LQ / 16), 256, 0, stream>>>(qp, kp, mask, out);
}
